// HGCN_38362647888412
// MI455X (gfx1250) — matrix operand order repaired and checked by reading, not compiled, not run on hardware
//
#include <hip/hip_runtime.h>
#include <hip/hip_bf16.h>
#include <cstdint>

// ---------------------------------------------------------------------------
// HGCN forward for MI455X (gfx1250, wave32). All GEMM-shaped stages run on
// v_wmma_f32_16x16x32_f16 (f16 in / f32 accumulate) with fused BN+activation
// epilogues; irregular stages (top-k, FPS, gathers, argmin) are VALU kernels.
// Each GEMM wave computes a 16x64 output strip (4 WMMA tiles) to reuse the
// A fragment 4x per K-step.
// ---------------------------------------------------------------------------

#define TPB 256
#define INVBN 0.9999950000374997f   // 1/sqrt(1 + 1e-5)
#define NEG_INF (-3.4e38f)

typedef __attribute__((ext_vector_type(16))) _Float16 v16h;
typedef __attribute__((ext_vector_type(8)))  float    v8f;

union FragU { v16h v; unsigned u[8]; };

static inline int ceil_div(int a, int b) { return (a + b - 1) / b; }

// ------------------------- (B,3,N) -> (B,N,3) ------------------------------
__global__ void k_transpose(const float* __restrict__ x, float* __restrict__ pc,
                            int B, int N) {
  int gid = blockIdx.x * blockDim.x + threadIdx.x;
  int total = B * N * 3;
  if (gid >= total) return;
  int c = gid % 3;
  int n = (gid / 3) % N;
  int b = gid / (3 * N);
  pc[gid] = x[((size_t)b * 3 + c) * N + n];
}

// ------------------ kNN (k=20) on 3-d coords, on-the-fly -------------------
__global__ void k_knn3(const float* __restrict__ p, int* __restrict__ idx,
                       int B, int N) {
  int gid = blockIdx.x * blockDim.x + threadIdx.x;
  if (gid >= B * N) return;
  int n = gid % N, b = gid / N;
  const float* pb = p + (size_t)b * N * 3;
  float cx = pb[n * 3], cy = pb[n * 3 + 1], cz = pb[n * 3 + 2];
  float tv[20]; int ti[20];
#pragma unroll
  for (int t = 0; t < 20; ++t) { tv[t] = NEG_INF; ti[t] = 0; }
  for (int j = 0; j < N; ++j) {
    float dx = pb[j * 3] - cx, dy = pb[j * 3 + 1] - cy, dz = pb[j * 3 + 2] - cz;
    float v = -(dx * dx + dy * dy + dz * dz);
    if (v > tv[19]) {
      int t = 19;
      while (t > 0 && v > tv[t - 1]) { tv[t] = tv[t - 1]; ti[t] = ti[t - 1]; --t; }
      tv[t] = v; ti[t] = j;
    }
  }
  int* o = idx + (size_t)gid * 20;
#pragma unroll
  for (int t = 0; t < 20; ++t) o[t] = ti[t];
}

// ------------- edge conv: out[b,n,o] = max_k leaky(bn(W . edge)) -----------
template <int C>
__global__ void k_edgeconv(const float* __restrict__ X, const int* __restrict__ idx,
                           const float* __restrict__ Wc, const float* __restrict__ gg,
                           const float* __restrict__ bb, float* __restrict__ out,
                           int B, int N) {
  const int O = 64, Kn = 20;
  int gid = blockIdx.x * blockDim.x + threadIdx.x;
  if (gid >= B * N * O) return;
  int o = gid % O;
  int n = (gid / O) % N;
  int b = gid / (O * N);
  const float* xi = X + ((size_t)b * N + n) * C;
  const float* w = Wc + (size_t)o * 2 * C;
  float xl[C];
  float cterm = 0.f;
  for (int c = 0; c < C; ++c) { xl[c] = xi[c]; cterm += w[C + c] * xl[c]; }
  const int* ids = idx + ((size_t)b * N + n) * Kn;
  float sc = gg[o] * INVBN, sb = bb[o];
  float best = NEG_INF;
  for (int kk = 0; kk < Kn; ++kk) {
    const float* xj = X + ((size_t)b * N + ids[kk]) * C;
    float s = cterm;
    for (int c = 0; c < C; ++c) s += w[c] * (xj[c] - xl[c]);
    s = s * sc + sb;
    s = (s >= 0.f) ? s : 0.2f * s;
    best = fmaxf(best, s);
  }
  out[gid] = best;
}

// -------------------- f32 -> f16 with optional K padding -------------------
__global__ void k_cvt_pad_f16(const float* __restrict__ src, _Float16* __restrict__ dst,
                              int R, int Cin, int Cpad) {
  int gid = blockIdx.x * blockDim.x + threadIdx.x;
  if (gid >= R * Cpad) return;
  int c = gid % Cpad;
  int r = gid / Cpad;
  float v = (c < Cin) ? src[(size_t)r * Cin + c] : 0.f;
  dst[gid] = (_Float16)v;
}

// --------------------------- WMMA GEMM (wave32) ----------------------------
// D[M,N] = A[M,K](row f16) * W[N,K](row f16)^T ; optional BN+act epilogue.
// One 16x64 strip per wave (4 WMMA tiles sharing the A fragment), 4 waves per
// block. K % 32 == 0, M % 16 == 0.
__global__ void k_gemm_wmma(const _Float16* __restrict__ A, int lda,
                            const _Float16* __restrict__ W, int ldw,
                            void* __restrict__ Dout, int ldd,
                            int M, int Nreal, int K,
                            int act, const float* __restrict__ gg,
                            const float* __restrict__ bbias,
                            float slope, int outHalf) {
  const int lane = threadIdx.x & 31;
  const int wave = threadIdx.x >> 5;
  const int mtile = blockIdx.x;
  const int nquad = blockIdx.y * 4 + wave;   // 64-column strip index
  const int half = lane >> 4;
  const int mr = lane & 15;                  // A row within tile
  const int nc = lane & 15;                  // B/D column within tile
  const int row0 = mtile * 16;
  const int ncolBase = nquad * 64;
  if (ncolBase >= Nreal) return;             // wave-uniform: whole strip OOB

  int ncol[4]; bool nvalid[4]; unsigned vmask[4];
#pragma unroll
  for (int q = 0; q < 4; ++q) {
    ncol[q] = ncolBase + q * 16 + nc;
    nvalid[q] = (ncol[q] < Nreal);
    vmask[q] = nvalid[q] ? 0xFFFFFFFFu : 0u;
  }
  const bool fullquad = (ncolBase + 64 <= Nreal);  // wave-uniform fast path

  v8f acc[4] = {};
  for (int k0 = 0; k0 < K; k0 += 32) {
    FragU a;
    // A 16x32 f16 fragment: lanes 0-15 hold K [0..15], lanes 16-31
    // hold [16..31] (two f16 per VGPR).
    const _Float16* arow = A + (size_t)(row0 + mr) * lda + k0 + half * 16;
#pragma unroll
    for (int i = 0; i < 4; ++i) a.u[i] = *(const unsigned*)(arow + 2 * i);
#pragma unroll
    for (int i = 0; i < 4; ++i) a.u[4 + i] = *(const unsigned*)(arow + 8 + 2 * i);
    if (k0 + 32 < K) __builtin_prefetch((const void*)(arow + 32), 0, 1);

    // B 32x16 fragments: lane = output column; lanes 0-15 K=0..15, 16-31 K=16..31.
    if (fullquad) {
#pragma unroll
      for (int q = 0; q < 4; ++q) {
        FragU bfr;
        const _Float16* wrow = W + (size_t)ncol[q] * ldw + k0 + half * 16;
#pragma unroll
        for (int j = 0; j < 8; ++j) bfr.u[j] = *(const unsigned*)(wrow + 2 * j);
        acc[q] = __builtin_amdgcn_wmma_f32_16x16x32_f16(false, a.v, false, bfr.v,
                                                        (short)0, acc[q], false, false);
      }
    } else {
#pragma unroll
      for (int q = 0; q < 4; ++q) {
        FragU bfr;
        int safec = nvalid[q] ? ncol[q] : 0;          // clamped row, no divergence
        const _Float16* wrow = W + (size_t)safec * ldw + k0 + half * 16;
#pragma unroll
        for (int j = 0; j < 8; ++j)
          bfr.u[j] = (*(const unsigned*)(wrow + 2 * j)) & vmask[q];
        acc[q] = __builtin_amdgcn_wmma_f32_16x16x32_f16(false, a.v, false, bfr.v,
                                                        (short)0, acc[q], false, false);
      }
    }
  }

#pragma unroll
  for (int q = 0; q < 4; ++q) {
    if (!nvalid[q]) continue;
    float sc = 1.f, sb = 0.f;
    if (act) { sc = gg[ncol[q]] * INVBN; sb = bbias[ncol[q]]; }
#pragma unroll
    for (int r = 0; r < 8; ++r) {
      int row = row0 + half * 8 + r;   // D layout: VGPR r -> M = 8*half + r
      float v = acc[q][r];
      if (act) {
        v = v * sc + sb;
        v = (act == 1) ? ((v >= 0.f) ? v : slope * v) : fmaxf(v, 0.f);
      }
      if (outHalf) ((_Float16*)Dout)[(size_t)row * ldd + ncol[q]] = (_Float16)v;
      else         ((float*)Dout)[(size_t)row * ldd + ncol[q]] = v;
    }
  }
}

// --------------------- Gram diag + kNN(20) from Gram -----------------------
__global__ void k_diag(const float* __restrict__ g, float* __restrict__ d, int N) {
  int i = blockIdx.x * blockDim.x + threadIdx.x;
  if (i < N) d[i] = g[(size_t)i * N + i];
}

__global__ void k_knn_gram(const float* __restrict__ G, const float* __restrict__ diag,
                           int* __restrict__ idx, int N) {
  int i = blockIdx.x * blockDim.x + threadIdx.x;
  if (i >= N) return;
  float gii = diag[i];
  const float* gr = G + (size_t)i * N;
  float tv[20]; int ti[20];
#pragma unroll
  for (int t = 0; t < 20; ++t) { tv[t] = NEG_INF; ti[t] = 0; }
  for (int j = 0; j < N; ++j) {
    float v = 2.f * gr[j] - gii - diag[j];   // -||xi-xj||^2
    if (v > tv[19]) {
      int t = 19;
      while (t > 0 && v > tv[t - 1]) { tv[t] = tv[t - 1]; ti[t] = ti[t - 1]; --t; }
      tv[t] = v; ti[t] = j;
    }
  }
  int* o = idx + (size_t)i * 20;
#pragma unroll
  for (int t = 0; t < 20; ++t) o[t] = ti[t];
}

// --------------------- farthest point sampling (per batch) -----------------
__global__ void k_fps(const float* __restrict__ P, int* __restrict__ smp, int N, int S) {
  int b = blockIdx.x, tid = threadIdx.x;
  __shared__ float dist[2048];
  __shared__ float rv[256];
  __shared__ int ri[256];
  __shared__ int far;
  const float* pb = P + (size_t)b * N * 3;
  for (int n = tid; n < N; n += 256) dist[n] = 1e10f;
  if (tid == 0) far = 0;
  __syncthreads();
  for (int s = 0; s < S; ++s) {
    int f = far;
    if (tid == 0) smp[b * S + s] = f;
    float cx = pb[f * 3], cy = pb[f * 3 + 1], cz = pb[f * 3 + 2];
    float bv = NEG_INF; int bi = 0;
    for (int n = tid; n < N; n += 256) {
      float dx = pb[n * 3] - cx, dy = pb[n * 3 + 1] - cy, dz = pb[n * 3 + 2] - cz;
      float d = dx * dx + dy * dy + dz * dz;
      float nd = fminf(dist[n], d);
      dist[n] = nd;
      if (nd > bv) { bv = nd; bi = n; }
    }
    rv[tid] = bv; ri[tid] = bi;
    __syncthreads();
    for (int off = 128; off > 0; off >>= 1) {
      if (tid < off) {
        if (rv[tid + off] > rv[tid] ||
            (rv[tid + off] == rv[tid] && ri[tid + off] < ri[tid])) {
          rv[tid] = rv[tid + off]; ri[tid] = ri[tid + off];
        }
      }
      __syncthreads();
    }
    if (tid == 0) far = ri[0];
    __syncthreads();
  }
}

// -------------------------- gather sampled coords --------------------------
__global__ void k_gather_p(const float* __restrict__ P, const int* __restrict__ s,
                           float* __restrict__ out, int B, int N, int Gn) {
  int gid = blockIdx.x * blockDim.x + threadIdx.x;
  if (gid >= B * Gn * 3) return;
  int c = gid % 3;
  int g = (gid / 3) % Gn;
  int b = gid / (3 * Gn);
  out[gid] = P[((size_t)b * N + s[b * Gn + g]) * 3 + c];
}

// ------------- per-group top-S nearest points (by 3-d distance) ------------
template <int S>
__global__ void k_group_knn(const float* __restrict__ P, const int* __restrict__ smp,
                            int* __restrict__ gi, int B, int N, int Gn) {
  int gid = blockIdx.x * blockDim.x + threadIdx.x;
  if (gid >= B * Gn) return;
  int b = gid / Gn;
  int ci = smp[gid];
  const float* pb = P + (size_t)b * N * 3;
  float cx = pb[ci * 3], cy = pb[ci * 3 + 1], cz = pb[ci * 3 + 2];
  float tv[S]; int ti[S];
#pragma unroll
  for (int t = 0; t < S; ++t) { tv[t] = NEG_INF; ti[t] = 0; }
  for (int j = 0; j < N; ++j) {
    float dx = pb[j * 3] - cx, dy = pb[j * 3 + 1] - cy, dz = pb[j * 3 + 2] - cz;
    float v = -(dx * dx + dy * dy + dz * dz);
    if (v > tv[S - 1]) {
      int t = S - 1;
      while (t > 0 && v > tv[t - 1]) { tv[t] = tv[t - 1]; ti[t] = ti[t - 1]; --t; }
      tv[t] = v; ti[t] = j;
    }
  }
  int* o = gi + (size_t)gid * S;
#pragma unroll
  for (int t = 0; t < S; ++t) o[t] = ti[t];
}

// ------------------ gather feature rows into f16 GEMM input ----------------
__global__ void k_gather_f16(const float* __restrict__ F, const int* __restrict__ gi,
                             _Float16* __restrict__ h, int B, int Gn, int S, int NF, int C) {
  int gid = blockIdx.x * blockDim.x + threadIdx.x;
  if (gid >= B * Gn * S * C) return;
  int c = gid % C;
  int s = (gid / C) % S;
  int g = (gid / (C * S)) % Gn;
  int b = gid / (C * S * Gn);
  int row = gi[((size_t)b * Gn + g) * S + s];
  h[gid] = (_Float16)F[((size_t)b * NF + row) * C + c];
}

// --------------------------- max over group dim ----------------------------
__global__ void k_max_s(const float* __restrict__ y, float* __restrict__ out,
                        int B, int Gn, int S, int H) {
  int gid = blockIdx.x * blockDim.x + threadIdx.x;
  if (gid >= B * Gn * H) return;
  int h = gid % H;
  int g = (gid / H) % Gn;
  int b = gid / (H * Gn);
  const float* base = y + (((size_t)b * Gn + g) * S) * H + h;
  float best = NEG_INF;
  for (int s = 0; s < S; ++s) best = fmaxf(best, base[(size_t)s * H]);
  out[gid] = best;
}

// ------------------ GCN chain pieces: column mean + vec x mat --------------
__global__ void k_colmean(const float* __restrict__ f, float* __restrict__ out,
                          int B, int Gn, int C) {
  int gid = blockIdx.x * blockDim.x + threadIdx.x;
  if (gid >= B * C) return;
  int c = gid % C, b = gid / C;
  float s = 0.f;
  for (int g = 0; g < Gn; ++g) s += f[((size_t)b * Gn + g) * C + c];
  out[gid] = s / (float)Gn;
}

__global__ void k_vecmat(const float* __restrict__ vin, const float* __restrict__ w,
                         const float* __restrict__ bias, float* __restrict__ vout,
                         int B, int Cin, int Cout, int ostride, int ooff) {
  int gid = blockIdx.x * blockDim.x + threadIdx.x;
  if (gid >= B * Cout) return;
  int o = gid % Cout, b = gid / Cout;
  float s = bias[o];
  const float* vi = vin + (size_t)b * Cin;
  for (int c = 0; c < Cin; ++c) s += vi[c] * w[(size_t)c * Cout + o];
  vout[(size_t)b * ostride + ooff + o] = s;
}

// ----------------------- nearest group index (argmin) ----------------------
__global__ void k_nearest(const float* __restrict__ P, const float* __restrict__ Q,
                          int* __restrict__ m, int B, int N, int Gn) {
  int gid = blockIdx.x * blockDim.x + threadIdx.x;
  if (gid >= B * N) return;
  int n = gid % N, b = gid / N;
  const float* pi = P + ((size_t)b * N + n) * 3;
  float px = pi[0], py = pi[1], pz = pi[2];
  const float* qb = Q + (size_t)b * Gn * 3;
  float best = 3.4e38f; int bi = 0;
  for (int g = 0; g < Gn; ++g) {
    float dx = px - qb[g * 3], dy = py - qb[g * 3 + 1], dz = pz - qb[g * 3 + 2];
    float d = dx * dx + dy * dy + dz * dz;
    if (d < best) { best = d; bi = g; }
  }
  m[gid] = bi;
}

// ---------------- assemble z = [loc | glob | onehot | pad] as f16 ----------
__global__ void k_build_z(const float* __restrict__ f0, const float* __restrict__ f1,
                          const float* __restrict__ f2, const float* __restrict__ glob,
                          const int* __restrict__ m0, const int* __restrict__ m1,
                          const int* __restrict__ m2, const int* __restrict__ label,
                          _Float16* __restrict__ zh, int B, int N) {
  const int CZ = 1248;
  long long gid = (long long)blockIdx.x * blockDim.x + threadIdx.x;
  long long total = (long long)B * N * CZ;
  if (gid >= total) return;
  int c = (int)(gid % CZ);
  int bn = (int)(gid / CZ);
  int b = bn / N;
  float v;
  if (c < 64)        v = f0[((size_t)b * N + m0[bn]) * 64 + c];
  else if (c < 192)  v = f1[((size_t)b * 32 + m1[bn]) * 128 + (c - 64)];
  else if (c < 448)  v = f2[((size_t)b * 8 + m2[bn]) * 256 + (c - 192)];
  else if (c < 1216) v = glob[(size_t)b * 768 + (c - 448)];
  else if (c < 1232) v = (label[b] == (c - 1216)) ? 1.f : 0.f;
  else               v = 0.f;
  zh[gid] = (_Float16)v;
}

// ---------------------------------------------------------------------------
static inline void launch_gemm(const _Float16* A, int lda, const _Float16* W, int ldw,
                               void* D, int ldd, int M, int Nreal, int K,
                               int act, const float* g, const float* b, float slope,
                               int outHalf, hipStream_t stream) {
  int ntiles = ceil_div(Nreal, 16);
  dim3 grid(M / 16, ceil_div(ntiles, 16));   // 4 waves x 4 tiles = 16 tiles/block
  k_gemm_wmma<<<grid, 128, 0, stream>>>(A, lda, W, ldw, D, ldd, M, Nreal, K,
                                        act, g, b, slope, outHalf);
}

extern "C" void kernel_launch(void* const* d_in, const int* in_sizes, int n_in,
                              void* d_out, int out_size, void* d_ws, size_t ws_size,
                              hipStream_t stream) {
  (void)in_sizes; (void)n_in; (void)out_size; (void)ws_size;
  const int B = 16, N = 2048, G1 = 32, S1 = 32, G2 = 8, S2 = 8;

  // ---- inputs (setup_inputs dict order, params flattened in insertion order)
  const float* X       = (const float*)d_in[0];
  const int*   label   = (const int*)d_in[1];
  const float* conv1_w = (const float*)d_in[2];
  const float* bn1_g   = (const float*)d_in[3];
  const float* bn1_b   = (const float*)d_in[4];
  const float* conv2_w = (const float*)d_in[5];
  const float* bn2_g   = (const float*)d_in[6];
  const float* bn2_b   = (const float*)d_in[7];
  const float* le1_w1  = (const float*)d_in[8];
  const float* le1_g1  = (const float*)d_in[9];
  const float* le1_b1  = (const float*)d_in[10];
  const float* le1_w2  = (const float*)d_in[11];
  const float* le1_g2  = (const float*)d_in[12];
  const float* le1_b2  = (const float*)d_in[13];
  const float* le2_w1  = (const float*)d_in[14];
  const float* le2_g1  = (const float*)d_in[15];
  const float* le2_b1  = (const float*)d_in[16];
  const float* le2_w2  = (const float*)d_in[17];
  const float* le2_g2  = (const float*)d_in[18];
  const float* le2_b2  = (const float*)d_in[19];
  const float* ge1_w1  = (const float*)d_in[20];
  const float* ge1_b1  = (const float*)d_in[21];
  const float* ge1_w2  = (const float*)d_in[22];
  const float* ge1_b2  = (const float*)d_in[23];
  const float* ge1_w3  = (const float*)d_in[24];
  const float* ge1_b3  = (const float*)d_in[25];
  const float* ge2_w1  = (const float*)d_in[26];
  const float* ge2_b1  = (const float*)d_in[27];
  const float* ge2_w2  = (const float*)d_in[28];
  const float* ge2_b2  = (const float*)d_in[29];
  const float* ge2_w3  = (const float*)d_in[30];
  const float* ge2_b3  = (const float*)d_in[31];
  const float* pc_w1   = (const float*)d_in[32];
  const float* pc_g1   = (const float*)d_in[33];
  const float* pc_b1   = (const float*)d_in[34];
  const float* pc_w2   = (const float*)d_in[35];
  const float* pc_g2   = (const float*)d_in[36];
  const float* pc_b2   = (const float*)d_in[37];
  const float* pc_w3   = (const float*)d_in[38];
  const float* pc_g3   = (const float*)d_in[39];
  const float* pc_b3   = (const float*)d_in[40];
  const float* pc_w4   = (const float*)d_in[41];

  // ---- workspace bump allocator
  char* base = (char*)d_ws;
  size_t off = 0;
  auto alloc = [&](size_t bytes) -> char* {
    char* p = base + off;
    off += (bytes + 255) & ~(size_t)255;
    return p;
  };

  // Persistent through z assembly / MLP
  float*     f1     = (float*)alloc((size_t)B * G1 * 128 * 4);
  float*     f2     = (float*)alloc((size_t)B * G2 * 256 * 4);
  float*     glob   = (float*)alloc((size_t)B * 768 * 4);
  int*       m0     = (int*)alloc((size_t)B * N * 4);
  int*       m1     = (int*)alloc((size_t)B * N * 4);
  int*       m2     = (int*)alloc((size_t)B * N * 4);
  _Float16*  zh     = (_Float16*)alloc((size_t)B * N * 1248 * 2);
  _Float16*  w1h    = (_Float16*)alloc((size_t)512 * 1248 * 2);
  _Float16*  w2h    = (_Float16*)alloc((size_t)256 * 512 * 2);
  _Float16*  w3h    = (_Float16*)alloc((size_t)128 * 256 * 2);
  _Float16*  w4h    = (_Float16*)alloc((size_t)50 * 128 * 2);
  _Float16*  lw11   = (_Float16*)alloc((size_t)128 * 64 * 2);
  _Float16*  lw12   = (_Float16*)alloc((size_t)128 * 128 * 2);
  _Float16*  lw21   = (_Float16*)alloc((size_t)256 * 128 * 2);
  _Float16*  lw22   = (_Float16*)alloc((size_t)256 * 256 * 2);
  int*       smp1   = (int*)alloc((size_t)B * G1 * 4);
  float*     pc1    = (float*)alloc((size_t)B * G1 * 3 * 4);
  int*       smp2   = (int*)alloc((size_t)B * G2 * 4);
  float*     pc2    = (float*)alloc((size_t)B * G2 * 3 * 4);
  int*       gidx1  = (int*)alloc((size_t)B * G1 * S1 * 4);
  int*       gidx2  = (int*)alloc((size_t)B * G2 * S2 * 4);
  float*     geA    = (float*)alloc((size_t)B * 512 * 4);
  float*     geB    = (float*)alloc((size_t)B * 512 * 4);
  float*     diag   = (float*)alloc((size_t)N * 4);

  // Phase-A scratch (reused by MLP temporaries after z is assembled)
  size_t scratch0 = off;
  float*     pc0   = (float*)alloc((size_t)B * N * 3 * 4);
  int*       idx1  = (int*)alloc((size_t)B * N * 20 * 4);
  int*       idx2  = (int*)alloc((size_t)B * N * 20 * 4);
  float*     x1    = (float*)alloc((size_t)B * N * 64 * 4);
  _Float16*  x1h   = (_Float16*)alloc((size_t)B * N * 64 * 2);
  float*     f0    = (float*)alloc((size_t)B * N * 64 * 4);
  float*     gram  = (float*)alloc((size_t)N * N * 4);
  _Float16*  h1h   = (_Float16*)alloc((size_t)B * G1 * S1 * 64 * 2);
  _Float16*  y1ha  = (_Float16*)alloc((size_t)B * G1 * S1 * 128 * 2);
  float*     y2a   = (float*)alloc((size_t)B * G1 * S1 * 128 * 4);
  _Float16*  h2h   = (_Float16*)alloc((size_t)B * G2 * S2 * 128 * 2);
  _Float16*  y1hb  = (_Float16*)alloc((size_t)B * G2 * S2 * 256 * 2);
  float*     y2b   = (float*)alloc((size_t)B * G2 * S2 * 256 * 4);

  // MLP f16 temporaries overlay phase-A scratch (stream-ordered; only written
  // after the last reader of the phase-A buffers has executed).
  off = scratch0;
  _Float16*  z1h = (_Float16*)alloc((size_t)B * N * 512 * 2);
  _Float16*  z2h = (_Float16*)alloc((size_t)B * N * 256 * 2);
  _Float16*  z3h = (_Float16*)alloc((size_t)B * N * 128 * 2);

  auto grid1 = [](int total) { return dim3((unsigned)ceil_div(total, TPB)); };

  // ---- weight conversions (f16, K padded to multiple of 32 where needed)
  k_cvt_pad_f16<<<grid1(512 * 1248), TPB, 0, stream>>>(pc_w1, w1h, 512, 1232, 1248);
  k_cvt_pad_f16<<<grid1(256 * 512), TPB, 0, stream>>>(pc_w2, w2h, 256, 512, 512);
  k_cvt_pad_f16<<<grid1(128 * 256), TPB, 0, stream>>>(pc_w3, w3h, 128, 256, 256);
  k_cvt_pad_f16<<<grid1(50 * 128), TPB, 0, stream>>>(pc_w4, w4h, 50, 128, 128);
  k_cvt_pad_f16<<<grid1(128 * 64), TPB, 0, stream>>>(le1_w1, lw11, 128, 64, 64);
  k_cvt_pad_f16<<<grid1(128 * 128), TPB, 0, stream>>>(le1_w2, lw12, 128, 128, 128);
  k_cvt_pad_f16<<<grid1(256 * 128), TPB, 0, stream>>>(le2_w1, lw21, 256, 128, 128);
  k_cvt_pad_f16<<<grid1(256 * 256), TPB, 0, stream>>>(le2_w2, lw22, 256, 256, 256);

  // ---- stage 1: transpose, kNN on coords, edge conv 1
  k_transpose<<<grid1(B * N * 3), TPB, 0, stream>>>(X, pc0, B, N);
  k_knn3<<<grid1(B * N), TPB, 0, stream>>>(pc0, idx1, B, N);
  k_edgeconv<3><<<grid1(B * N * 64), TPB, 0, stream>>>(pc0, idx1, conv1_w, bn1_g,
                                                       bn1_b, x1, B, N);

  // ---- stage 2: kNN on 64-d features via per-batch WMMA Gram, edge conv 2
  k_cvt_pad_f16<<<grid1(B * N * 64), TPB, 0, stream>>>(x1, x1h, B * N, 64, 64);
  for (int b = 0; b < B; ++b) {
    const _Float16* xb = x1h + (size_t)b * N * 64;
    launch_gemm(xb, 64, xb, 64, gram, N, N, N, 64, 0, nullptr, nullptr, 0.f, 0, stream);
    k_diag<<<grid1(N), TPB, 0, stream>>>(gram, diag, N);
    k_knn_gram<<<grid1(N), TPB, 0, stream>>>(gram, diag, idx2 + (size_t)b * N * 20, N);
  }
  k_edgeconv<64><<<grid1(B * N * 64), TPB, 0, stream>>>(x1, idx2, conv2_w, bn2_g,
                                                        bn2_b, f0, B, N);

  // ---- hierarchy level 1: FPS(32) -> groups(32) -> local embedder (WMMA)
  k_fps<<<B, 256, 0, stream>>>(pc0, smp1, N, G1);
  k_gather_p<<<grid1(B * G1 * 3), TPB, 0, stream>>>(pc0, smp1, pc1, B, N, G1);
  k_group_knn<S1><<<grid1(B * G1), TPB, 0, stream>>>(pc0, smp1, gidx1, B, N, G1);
  k_gather_f16<<<grid1(B * G1 * S1 * 64), TPB, 0, stream>>>(f0, gidx1, h1h, B, G1, S1, N, 64);
  launch_gemm(h1h, 64, lw11, 64, y1ha, 128, B * G1 * S1, 128, 64,
              2, le1_g1, le1_b1, 0.f, 1, stream);
  launch_gemm(y1ha, 128, lw12, 128, y2a, 128, B * G1 * S1, 128, 128,
              2, le1_g2, le1_b2, 0.f, 0, stream);
  k_max_s<<<grid1(B * G1 * 128), TPB, 0, stream>>>(y2a, f1, B, G1, S1, 128);

  // ---- hierarchy level 2: FPS(8) -> groups(8) -> local embedder (WMMA)
  k_fps<<<B, 256, 0, stream>>>(pc1, smp2, G1, G2);
  k_gather_p<<<grid1(B * G2 * 3), TPB, 0, stream>>>(pc1, smp2, pc2, B, G1, G2);
  k_group_knn<S2><<<grid1(B * G2), TPB, 0, stream>>>(pc1, smp2, gidx2, B, G1, G2);
  k_gather_f16<<<grid1(B * G2 * S2 * 128), TPB, 0, stream>>>(f1, gidx2, h2h, B, G2, S2, G1, 128);
  launch_gemm(h2h, 128, lw21, 128, y1hb, 256, B * G2 * S2, 256, 128,
              2, le2_g1, le2_b1, 0.f, 1, stream);
  launch_gemm(y1hb, 256, lw22, 256, y2b, 256, B * G2 * S2, 256, 256,
              2, le2_g2, le2_b2, 0.f, 0, stream);
  k_max_s<<<grid1(B * G2 * 256), TPB, 0, stream>>>(y2b, f2, B, G2, S2, 256);

  // ---- graph embedders (GCN on complete graph == chained mean @ W + b)
  k_colmean<<<grid1(B * 128), TPB, 0, stream>>>(f1, geA, B, G1, 128);
  k_vecmat<<<grid1(B * 256), TPB, 0, stream>>>(geA, ge1_w1, ge1_b1, geB, B, 128, 256, 256, 0);
  k_vecmat<<<grid1(B * 256), TPB, 0, stream>>>(geB, ge1_w2, ge1_b2, geA, B, 256, 256, 256, 0);
  k_vecmat<<<grid1(B * 256), TPB, 0, stream>>>(geA, ge1_w3, ge1_b3, glob, B, 256, 256, 768, 0);
  k_colmean<<<grid1(B * 256), TPB, 0, stream>>>(f2, geA, B, G2, 256);
  k_vecmat<<<grid1(B * 512), TPB, 0, stream>>>(geA, ge2_w1, ge2_b1, geB, B, 256, 512, 512, 0);
  k_vecmat<<<grid1(B * 512), TPB, 0, stream>>>(geB, ge2_w2, ge2_b2, geA, B, 512, 512, 512, 0);
  k_vecmat<<<grid1(B * 512), TPB, 0, stream>>>(geA, ge2_w3, ge2_b3, glob, B, 512, 512, 768, 256);

  // ---- nearest-group indices + z assembly
  k_nearest<<<grid1(B * N), TPB, 0, stream>>>(pc0, pc0, m0, B, N, N);
  k_nearest<<<grid1(B * N), TPB, 0, stream>>>(pc0, pc1, m1, B, N, G1);
  k_nearest<<<grid1(B * N), TPB, 0, stream>>>(pc0, pc2, m2, B, N, G2);
  {
    long long total = (long long)B * N * 1248;
    dim3 g((unsigned)((total + TPB - 1) / TPB));
    k_build_z<<<g, TPB, 0, stream>>>(f0, f1, f2, glob, m0, m1, m2, label, zh, B, N);
  }

  // ---- final per-point MLP, all WMMA with fused BN + leaky(0.1)
  launch_gemm(zh, 1248, w1h, 1248, z1h, 512, B * N, 512, 1248,
              1, pc_g1, pc_b1, 0.1f, 1, stream);
  launch_gemm(z1h, 512, w2h, 512, z2h, 256, B * N, 256, 512,
              1, pc_g2, pc_b2, 0.1f, 1, stream);
  launch_gemm(z2h, 256, w3h, 256, z3h, 128, B * N, 128, 256,
              1, pc_g3, pc_b3, 0.1f, 1, stream);
  launch_gemm(z3h, 128, w4h, 128, d_out, 50, B * N, 50, 128,
              0, nullptr, nullptr, 0.f, 0, stream);
}
